// IMU_LSTM_7035156431314
// MI455X (gfx1250) — matrix operand order repaired and checked by reading, not compiled, not run on hardware
//
#include <hip/hip_runtime.h>
#include <math.h>

// ---------------------------------------------------------------------------
// Problem constants (from reference): B=1024, T=512, D=6, H=128, OUT=8
// ---------------------------------------------------------------------------
#define BB 1024
#define TT 512
#define DD 6
#define HH 128
#define G4H 512   // 4*H
#define OUTD 8

typedef __attribute__((ext_vector_type(16))) __bf16 v16bf;
typedef __attribute__((ext_vector_type(8)))  float  v8f;
typedef __bf16 bf16;

// ---------------------------------------------------------------------------
// helpers
// ---------------------------------------------------------------------------
__device__ __forceinline__ bf16 f2bf(float x) { return (bf16)x; }
__device__ __forceinline__ float bf2f(bf16 x) { return (float)x; }

__device__ __forceinline__ float gelu_erf(float x) {
  return 0.5f * x * (1.0f + erff(x * 0.70710678118654752f));
}
__device__ __forceinline__ float sigmoid_f(float x) {
  return 1.0f / (1.0f + __expf(-x));
}

// WMMA wrapper: D = A(16x32 bf16) * B(32x16 bf16) + C(16x16 f32)
__device__ __forceinline__ v8f wmma_bf16(v16bf a, v16bf b, v8f c) {
  return __builtin_amdgcn_wmma_f32_16x16x32_bf16(
      /*neg_a=*/false, a, /*neg_b=*/false, b,
      /*c_mod=*/(short)0, c, /*reuse_a=*/false, /*reuse_b=*/false);
}

// Load A fragment (16x32, 16-bit) from row-major bf16 matrix, row stride ld.
// Per ISA 7.12.2: lanes 0-15 -> M=lane; elems 0..7 = K (hi*8)..(hi*8+7),
// elems 8..15 = K (16+hi*8).. ; hi = lane>>4.
__device__ __forceinline__ v16bf load_a(const bf16* p, int ld) {
  int lane = threadIdx.x & 31;
  int l = lane & 15, hi = lane >> 4;
  const bf16* r = p + (size_t)l * ld;
  union { uint4 u[2]; v16bf v; } t;
  t.u[0] = *reinterpret_cast<const uint4*>(r + hi * 8);
  t.u[1] = *reinterpret_cast<const uint4*>(r + 16 + hi * 8);
  return t.v;
}

// Load B fragment (32x16) representing B[k][n] = W[n][k] from row-major
// W (rows = output features n, cols = k), row stride ldw.
// n=lane&15; elems 0..7 = K (hi*8)..(hi*8+7), elems 8..15 = K (16+hi*8).. ; hi = lane>>4.
__device__ __forceinline__ v16bf load_b(const bf16* W, int ldw) {
  int lane = threadIdx.x & 31;
  int l = lane & 15, hi = lane >> 4;
  const bf16* r = W + (size_t)l * ldw;
  union { uint4 u[2]; v16bf v; } t;
  t.u[0] = *reinterpret_cast<const uint4*>(r + hi * 8);
  t.u[1] = *reinterpret_cast<const uint4*>(r + 16 + hi * 8);
  return t.v;
}

// ---------------------------------------------------------------------------
// tiny conversion / bias kernels
// ---------------------------------------------------------------------------
__global__ void cvt_f32_bf16_kernel(const float* __restrict__ in,
                                    bf16* __restrict__ out, int n) {
  int i = blockIdx.x * blockDim.x + threadIdx.x;
  if (i < n) out[i] = f2bf(in[i]);
}

__global__ void add_bias_kernel(const float* __restrict__ a,
                                const float* __restrict__ b,
                                float* __restrict__ out, int n) {
  int i = blockIdx.x * blockDim.x + threadIdx.x;
  if (i < n) out[i] = a[i] + b[i];
}

// ---------------------------------------------------------------------------
// One 128x128 @ 128x128^T GEMM layer + bias + gelu, LDS -> LDS.
// Called with the weight pointer passed DIRECTLY (no pointer array) so the
// loads keep addrspace(1) and lower to global_load_b128, not flat_load.
// ---------------------------------------------------------------------------
__device__ __forceinline__ void mlp_layer_128(
    bf16 (*Xs)[136], const bf16* __restrict__ W, const float* __restrict__ bias,
    int wave, int l, int hi) {
  v8f acc[8];
#pragma unroll
  for (int nt = 0; nt < 8; ++nt)
#pragma unroll
    for (int i = 0; i < 8; ++i) acc[nt][i] = 0.0f;

#pragma unroll
  for (int kk = 0; kk < 4; ++kk) {
    v16bf a = load_a(&Xs[wave * 16][kk * 32], 136);
#pragma unroll
    for (int nt = 0; nt < 8; ++nt) {
      v16bf b = load_b(W + (size_t)(nt * 16) * HH + kk * 32, HH);
      acc[nt] = wmma_bf16(a, b, acc[nt]);
    }
  }
  __syncthreads();   // all reads of Xs done before overwrite
#pragma unroll
  for (int nt = 0; nt < 8; ++nt) {
    int n = nt * 16 + l;
    float bv = bias[n];
#pragma unroll
    for (int r = 0; r < 8; ++r) {
      int m = wave * 16 + hi * 8 + r;
      Xs[m][n] = f2bf(gelu_erf(acc[nt][r] + bv));
    }
  }
  __syncthreads();
}

// ---------------------------------------------------------------------------
// Fused pointwise conv stack:  src (B,T,6) -> gelu-MLP x4 -> +pos -> (T,B,H) bf16
// 4096 blocks x 256 threads; each block owns 128 rows of (B*T).
// ---------------------------------------------------------------------------
__global__ __launch_bounds__(256) void conv_stack_kernel(
    const float* __restrict__ src,
    const float* __restrict__ w0, const float* __restrict__ b0,
    const bf16*  __restrict__ w1b, const float* __restrict__ b1,
    const bf16*  __restrict__ w2b, const float* __restrict__ b2,
    const bf16*  __restrict__ w3b, const float* __restrict__ b3,
    const float* __restrict__ pos,
    bf16* __restrict__ xout) {
  __shared__ bf16 Xs[128][136];   // 128x128 tile, padded rows (272B, 16B aligned)

  const int tid = threadIdx.x;
  const int row0 = blockIdx.x * 128;   // base row in flattened (B*T)

  // ---- layer 0: D=6 -> H=128, gelu (scalar; K too small for WMMA) ----
  {
    int row = tid >> 1;                 // 2 threads per row
    int c0 = (tid & 1) * 64;
    const float* s = src + (size_t)(row0 + row) * DD;
    float sv[DD];
#pragma unroll
    for (int d = 0; d < DD; ++d) sv[d] = s[d];
    for (int n = c0; n < c0 + 64; ++n) {
      const float* w = w0 + n * DD;
      float acc = b0[n];
#pragma unroll
      for (int d = 0; d < DD; ++d) acc += sv[d] * w[d];
      Xs[row][n] = f2bf(gelu_erf(acc));
    }
  }
  __syncthreads();

  // ---- layers 1..3: WMMA GEMM + gelu, each with a direct weight pointer ----
  const int wave = tid >> 5;
  const int lane = tid & 31;
  const int l = lane & 15, hi = lane >> 4;
  mlp_layer_128(Xs, w1b, b1, wave, l, hi);
  mlp_layer_128(Xs, w2b, b2, wave, l, hi);
  mlp_layer_128(Xs, w3b, b3, wave, l, hi);

  // ---- add pos_embed, write time-major (T,B,H) bf16, coalesced per row ----
  for (int r = tid >> 6; r < 128; r += 4) {      // 4 rows in flight, 64 threads/row
    int n = (tid & 63) * 2;
    int g = row0 + r;
    int bidx = g >> 9;        // / T (=512)
    int t    = g & 511;       // % T
    const float* pp = pos + (size_t)t * HH;
    bf16* o = xout + ((size_t)t * BB + bidx) * HH;
    o[n]     = f2bf(bf2f(Xs[r][n])     + pp[n]);
    o[n + 1] = f2bf(bf2f(Xs[r][n + 1]) + pp[n + 1]);
  }
}

// ---------------------------------------------------------------------------
// One LSTM time step:  z = x@Wih^T (+ h@Whh^T) + bias ; gate math ; c,h update.
// Grid: 64 blocks x 256 threads = 512 waves = 64 M-tiles x 8 N-tiles(within H).
// Each wave owns one 16x16 tile for all four gate blocks (i,f,g,o in regs).
// ---------------------------------------------------------------------------
__global__ __launch_bounds__(256) void lstm_step_kernel(
    const bf16* __restrict__ xin,     // (B,H) layer input at time t
    const bf16* __restrict__ hprev,   // (B,H) previous hidden (unused if first)
    float* __restrict__ c,            // (B,H) cell state, in/out (in-place OK)
    const bf16* __restrict__ Wih,     // (4H,H) bf16
    const bf16* __restrict__ Whh,     // (4H,H) bf16
    const float* __restrict__ bias,   // (4H) combined bih+bhh
    bf16* __restrict__ hout,          // (B,H)
    int first) {
  const int wave = threadIdx.x >> 5;
  const int wg = blockIdx.x * 8 + wave;   // 0..511
  const int m0 = (wg >> 3) * 16;          // row tile in B
  const int n0 = (wg & 7) * 16;           // col tile in H
  const int lane = threadIdx.x & 31;
  const int l = lane & 15, hi = lane >> 4;

  v8f z[4];
#pragma unroll
  for (int g = 0; g < 4; ++g)
#pragma unroll
    for (int i = 0; i < 8; ++i) z[g][i] = 0.0f;

#pragma unroll
  for (int kk = 0; kk < 4; ++kk) {
    v16bf a = load_a(xin + (size_t)m0 * HH + kk * 32, HH);
#pragma unroll
    for (int g = 0; g < 4; ++g) {
      v16bf b = load_b(Wih + (size_t)(g * HH + n0) * HH + kk * 32, HH);
      z[g] = wmma_bf16(a, b, z[g]);
    }
  }
  if (!first) {
#pragma unroll
    for (int kk = 0; kk < 4; ++kk) {
      v16bf a = load_a(hprev + (size_t)m0 * HH + kk * 32, HH);
#pragma unroll
      for (int g = 0; g < 4; ++g) {
        v16bf b = load_b(Whh + (size_t)(g * HH + n0) * HH + kk * 32, HH);
        z[g] = wmma_bf16(a, b, z[g]);
      }
    }
  }

  const float bi = bias[0 * HH + n0 + l];
  const float bf_ = bias[1 * HH + n0 + l];
  const float bg = bias[2 * HH + n0 + l];
  const float bo = bias[3 * HH + n0 + l];
#pragma unroll
  for (int r = 0; r < 8; ++r) {
    int m = m0 + hi * 8 + r;
    size_t idx = (size_t)m * HH + n0 + l;
    float cprev = first ? 0.0f : c[idx];
    float iv = sigmoid_f(z[0][r] + bi);
    float fv = sigmoid_f(z[1][r] + bf_);
    float gv = tanhf(z[2][r] + bg);
    float ov = sigmoid_f(z[3][r] + bo);
    float cn = fv * cprev + iv * gv;
    float hn = ov * tanhf(cn);
    c[idx] = cn;
    hout[idx] = f2bf(hn);
  }
}

// ---------------------------------------------------------------------------
// Head: LayerNorm(eps=1e-5) -> Linear(128,32) -> GELU -> Linear(32,8)
// One thread per batch row (1024 threads).
// ---------------------------------------------------------------------------
__global__ __launch_bounds__(256) void head_kernel(
    const bf16* __restrict__ h,
    const float* __restrict__ ln_g, const float* __restrict__ ln_b,
    const float* __restrict__ hw1, const float* __restrict__ hb1,
    const float* __restrict__ hw2, const float* __restrict__ hb2,
    float* __restrict__ out) {
  int row = blockIdx.x * blockDim.x + threadIdx.x;
  if (row >= BB) return;
  float x[HH];
  float mu = 0.0f;
  for (int i = 0; i < HH; ++i) {
    x[i] = bf2f(h[(size_t)row * HH + i]);
    mu += x[i];
  }
  mu *= (1.0f / HH);
  float var = 0.0f;
  for (int i = 0; i < HH; ++i) {
    float d = x[i] - mu;
    var += d * d;
  }
  var *= (1.0f / HH);
  float rs = rsqrtf(var + 1e-5f);
  for (int i = 0; i < HH; ++i)
    x[i] = (x[i] - mu) * rs * ln_g[i] + ln_b[i];

  float y[HH / 4];
  for (int j = 0; j < HH / 4; ++j) {
    float acc = hb1[j];
    for (int i = 0; i < HH; ++i) acc += x[i] * hw1[(size_t)j * HH + i];
    y[j] = gelu_erf(acc);
  }
  for (int k = 0; k < OUTD; ++k) {
    float acc = hb2[k];
    for (int j = 0; j < HH / 4; ++j) acc += y[j] * hw2[(size_t)k * (HH / 4) + j];
    out[(size_t)row * OUTD + k] = acc;
  }
}

// ---------------------------------------------------------------------------
// host launcher
// ---------------------------------------------------------------------------
extern "C" void kernel_launch(void* const* d_in, const int* in_sizes, int n_in,
                              void* d_out, int out_size, void* d_ws, size_t ws_size,
                              hipStream_t stream) {
  const float* src  = (const float*)d_in[0];
  const float* w0   = (const float*)d_in[1];
  const float* b0   = (const float*)d_in[2];
  const float* w1   = (const float*)d_in[3];
  const float* b1   = (const float*)d_in[4];
  const float* w2   = (const float*)d_in[5];
  const float* b2   = (const float*)d_in[6];
  const float* w3   = (const float*)d_in[7];
  const float* b3   = (const float*)d_in[8];
  const float* pos  = (const float*)d_in[9];
  const float* Wih0 = (const float*)d_in[10];
  const float* Whh0 = (const float*)d_in[11];
  const float* bih0 = (const float*)d_in[12];
  const float* bhh0 = (const float*)d_in[13];
  const float* Wih1 = (const float*)d_in[14];
  const float* Whh1 = (const float*)d_in[15];
  const float* bih1 = (const float*)d_in[16];
  const float* bhh1 = (const float*)d_in[17];
  const float* ln_g = (const float*)d_in[18];
  const float* ln_b = (const float*)d_in[19];
  const float* hw1  = (const float*)d_in[20];
  const float* hb1  = (const float*)d_in[21];
  const float* hw2  = (const float*)d_in[22];
  const float* hb2  = (const float*)d_in[23];
  float* out = (float*)d_out;

  // ---- carve workspace (bump allocator, 256B aligned) ----
  char* ws = (char*)d_ws;
  size_t off = 0;
  auto take = [&](size_t bytes) -> void* {
    off = (off + 255) & ~(size_t)255;
    void* p = ws + off;
    off += bytes;
    return p;
  };
  const size_t TBH = (size_t)TT * BB * HH;
  bf16* xbf   = (bf16*)take(TBH * sizeof(bf16));          // conv-stack output, (T,B,H)
  bf16* ys0   = (bf16*)take(TBH * sizeof(bf16));          // layer-0 hidden sequence
  bf16* h1a   = (bf16*)take((size_t)BB * HH * sizeof(bf16));
  bf16* h1b   = (bf16*)take((size_t)BB * HH * sizeof(bf16));
  float* c0   = (float*)take((size_t)BB * HH * sizeof(float));
  float* c1   = (float*)take((size_t)BB * HH * sizeof(float));
  bf16* w1b   = (bf16*)take((size_t)HH * HH * sizeof(bf16));
  bf16* w2b   = (bf16*)take((size_t)HH * HH * sizeof(bf16));
  bf16* w3b   = (bf16*)take((size_t)HH * HH * sizeof(bf16));
  bf16* Wih0b = (bf16*)take((size_t)G4H * HH * sizeof(bf16));
  bf16* Whh0b = (bf16*)take((size_t)G4H * HH * sizeof(bf16));
  bf16* Wih1b = (bf16*)take((size_t)G4H * HH * sizeof(bf16));
  bf16* Whh1b = (bf16*)take((size_t)G4H * HH * sizeof(bf16));
  float* bias0 = (float*)take((size_t)G4H * sizeof(float));
  float* bias1 = (float*)take((size_t)G4H * sizeof(float));

  // ---- weight conversion + bias combine ----
  const int nHH = HH * HH, n4H = G4H * HH;
  cvt_f32_bf16_kernel<<<(nHH + 255) / 256, 256, 0, stream>>>(w1, w1b, nHH);
  cvt_f32_bf16_kernel<<<(nHH + 255) / 256, 256, 0, stream>>>(w2, w2b, nHH);
  cvt_f32_bf16_kernel<<<(nHH + 255) / 256, 256, 0, stream>>>(w3, w3b, nHH);
  cvt_f32_bf16_kernel<<<(n4H + 255) / 256, 256, 0, stream>>>(Wih0, Wih0b, n4H);
  cvt_f32_bf16_kernel<<<(n4H + 255) / 256, 256, 0, stream>>>(Whh0, Whh0b, n4H);
  cvt_f32_bf16_kernel<<<(n4H + 255) / 256, 256, 0, stream>>>(Wih1, Wih1b, n4H);
  cvt_f32_bf16_kernel<<<(n4H + 255) / 256, 256, 0, stream>>>(Whh1, Whh1b, n4H);
  add_bias_kernel<<<(G4H + 255) / 256, 256, 0, stream>>>(bih0, bhh0, bias0, G4H);
  add_bias_kernel<<<(G4H + 255) / 256, 256, 0, stream>>>(bih1, bhh1, bias1, G4H);

  // ---- fused pointwise conv stack + pos embed, time-major bf16 out ----
  conv_stack_kernel<<<(BB * TT) / 128, 256, 0, stream>>>(
      src, w0, b0, w1b, b1, w2b, b2, w3b, b3, pos, xbf);

  // ---- two-layer LSTM, one launch per (step, layer); stream order = dep order
  const size_t BH = (size_t)BB * HH;
  for (int t = 0; t < TT; ++t) {
    const bf16* h0prev = (t == 0) ? nullptr : (ys0 + (size_t)(t - 1) * BH);
    lstm_step_kernel<<<64, 256, 0, stream>>>(
        xbf + (size_t)t * BH, h0prev, c0, Wih0b, Whh0b, bias0,
        ys0 + (size_t)t * BH, t == 0 ? 1 : 0);
    bf16* h1in  = (t & 1) ? h1b : h1a;
    bf16* h1out = (t & 1) ? h1a : h1b;
    lstm_step_kernel<<<64, 256, 0, stream>>>(
        ys0 + (size_t)t * BH, h1in, c1, Wih1b, Whh1b, bias1, h1out, t == 0 ? 1 : 0);
  }
  // T=512 even -> final hidden ended up in h1a (written at t=511, t&1 -> out=h1a)
  head_kernel<<<(BB + 255) / 256, 256, 0, stream>>>(
      h1a, ln_g, ln_b, hw1, hb1, hw2, hb2, out);
}
